// DeepFM_74448963109120
// MI455X (gfx1250) — matrix operand order repaired and checked by reading, not compiled, not run on hardware
//
#include <hip/hip_runtime.h>
#include <hip/hip_bf16.h>
#include <math.h>

// ---------------- WMMA fragment types (gfx1250, wave32) ----------------
typedef __attribute__((ext_vector_type(16))) __bf16        v16bf;
typedef __attribute__((ext_vector_type(8)))  float         v8f;
typedef __attribute__((ext_vector_type(8)))  unsigned int  v8u;

// ---------------- Problem constants ----------------
#define NROWS   16384
#define NFEAT   39
#define VOCAB   100000
#define EDIM    16
#define HIDDEN  400
#define K1      624      // NFEAT*EDIM
#define K1P     640      // padded to 32
#define K2P     416      // 400 padded to 32
#define BNEPS   1e-5f

__device__ __forceinline__ unsigned short f2bf(float f) {
  unsigned int u = __float_as_uint(f);
  u += 0x7FFFu + ((u >> 16) & 1u);   // round-to-nearest-even
  return (unsigned short)(u >> 16);
}

// ---------------- 1) Weight fp32 -> bf16 with K zero-padding ----------------
__global__ __launch_bounds__(256) void convert_w_kernel(
    const float* __restrict__ W, unsigned short* __restrict__ Wb,
    int Nout, int K, int Kp)
{
  size_t i = (size_t)blockIdx.x * 256 + threadIdx.x;
  if (i >= (size_t)Nout * Kp) return;
  int col = (int)(i % Kp);
  int row = (int)(i / Kp);
  float v = (col < K) ? W[(size_t)row * K + col] : 0.f;
  Wb[i] = f2bf(v);
}

// ---------------- 2) Embedding gather + FM terms + bf16 h0 ----------------
// One wave32 per row. Lanes: e = lane&15 over EDIM, lane>>4 splits features.
__global__ __launch_bounds__(256) void embed_fm_kernel(
    const int*   __restrict__ Xi,   const float* __restrict__ Xv,
    const float* __restrict__ emb1, const float* __restrict__ emb2,
    unsigned short* __restrict__ h0, float* __restrict__ fm)
{
  __shared__ float xvbuf[8 * K1];
  const int lane = threadIdx.x & 31;
  const int wid  = threadIdx.x >> 5;
  const int n    = blockIdx.x * 8 + wid;
  float* xvrow   = xvbuf + wid * K1;
  const int e    = lane & 15;
  const int half = lane >> 4;

  float sumsq = 0.f, ssum = 0.f;
  for (int f = half; f < NFEAT; f += 2) {
    int   id  = Xi[n * NFEAT + f];
    float xvv = Xv[n * NFEAT + f];
    float v   = emb2[((size_t)f * VOCAB + id) * EDIM + e] * xvv;
    xvrow[f * EDIM + e] = v;
    sumsq += v * v;
    ssum  += v;
  }
  __syncthreads();   // all rows in block finished writing xv to LDS

  // combine the two feature halves (lane <-> lane^16)
  sumsq += __shfl_xor(sumsq, 16, 32);
  ssum  += __shfl_xor(ssum, 16, 32);

  float norm = fmaxf(sqrtf(sumsq), 1e-12f);
  float inv  = 1.0f / norm;
  float s    = ssum * inv;
  float f2   = 0.5f * (s * s - sumsq * inv * inv);
  float f2sum = f2;
  for (int off = 1; off < 16; off <<= 1) f2sum += __shfl_xor(f2sum, off, 32);

  // first-order term
  float f1 = 0.f;
  for (int f = lane; f < NFEAT; f += 32) {
    int id = Xi[n * NFEAT + f];
    f1 += emb1[(size_t)f * VOCAB + id] * Xv[n * NFEAT + f];
  }
  for (int off = 1; off < 32; off <<= 1) f1 += __shfl_xor(f1, off, 32);

  // normalized h row -> bf16 (padded to K1P)
  for (int i = lane; i < K1P; i += 32) {
    float hv = 0.f;
    if (i < K1) {
      float invi = __shfl(inv, i & 15, 32);
      hv = xvrow[i] * invi;
    }
    h0[(size_t)n * K1P + i] = f2bf(hv);
  }
  if (lane == 0) fm[n] = f1 + f2sum;
}

// ---------------- 3) GEMM: Z[M,Nout] = A[M,Kp](bf16) * W[Nout,Kp]^T + bias ----------------
// One 16x16 output tile per wave; v_wmma_f32_16x16x32_bf16 over the K loop.
__global__ __launch_bounds__(256) void gemm_wmma_kernel(
    const unsigned short* __restrict__ A, const unsigned short* __restrict__ W,
    const float* __restrict__ bias, float* __restrict__ Z,
    int M, int Nout, int Kp)
{
  const int lane  = threadIdx.x & 31;
  const int wid   = threadIdx.x >> 5;
  const int ntn   = Nout >> 4;
  const int gtile = blockIdx.x * 8 + wid;
  const int tm    = gtile / ntn;
  const int tn    = gtile - tm * ntn;
  const int m0    = tm << 4, n0 = tn << 4;

  const int r15 = lane & 15;
  const int hi  = lane >> 4;
  // A frag: lane(0-15)=row M; halves split K 0-15 / 16-31
  const unsigned short* arow = A + (size_t)(m0 + r15) * Kp + hi * 16;
  // B frag (=W^T): lane holds column n=r15; halves split K 0-15 / 16-31
  const unsigned short* brow = W + (size_t)(n0 + r15) * Kp + hi * 16;

  v8f acc = {};
  for (int k0 = 0; k0 < Kp; k0 += 32) {
    v8u au, bu;
#pragma unroll
    for (int v = 0; v < 8; ++v)
      au[v]     = *reinterpret_cast<const unsigned int*>(arow + k0 + 2 * v);
#pragma unroll
    for (int v = 0; v < 8; ++v)
      bu[v]     = *reinterpret_cast<const unsigned int*>(brow + k0 + 2 * v);

    acc = __builtin_amdgcn_wmma_f32_16x16x32_bf16(
        false, __builtin_bit_cast(v16bf, au),
        false, __builtin_bit_cast(v16bf, bu),
        (short)0, acc, false, false);
  }

  const int   ncol = n0 + r15;
  const float bn   = bias[ncol];
#pragma unroll
  for (int r = 0; r < 8; ++r) {
    int mrow = m0 + r + hi * 8;   // C/D layout: VGPR r -> M=r (lanes 0-15) / M=r+8
    Z[(size_t)mrow * Nout + ncol] = acc[r] + bn;
  }
}

// ---------------- 4) BatchNorm statistics: per-column scale/shift ----------------
__global__ __launch_bounds__(256) void bn_stats_kernel(
    const float* __restrict__ Z, const float* __restrict__ g, const float* __restrict__ bt,
    float* __restrict__ scale, float* __restrict__ shift, int M, int Nout)
{
  const int j = blockIdx.x;
  float s = 0.f, q = 0.f;
  for (int m = threadIdx.x; m < M; m += 256) {
    float z = Z[(size_t)m * Nout + j];
    s += z; q += z * z;
  }
  __shared__ float sb[256], qb[256];
  sb[threadIdx.x] = s; qb[threadIdx.x] = q;
  __syncthreads();
  for (int off = 128; off > 0; off >>= 1) {
    if (threadIdx.x < off) {
      sb[threadIdx.x] += sb[threadIdx.x + off];
      qb[threadIdx.x] += qb[threadIdx.x + off];
    }
    __syncthreads();
  }
  if (threadIdx.x == 0) {
    float mean = sb[0] / (float)M;
    float var  = qb[0] / (float)M - mean * mean;
    float sc   = g[j] * rsqrtf(var + BNEPS);
    scale[j] = sc;
    shift[j] = bt[j] - mean * sc;
  }
}

// ---------------- 5) BN + ReLU + requantize to padded bf16 ----------------
__global__ __launch_bounds__(256) void bn_relu_bf16_kernel(
    const float* __restrict__ Z, const float* __restrict__ scale, const float* __restrict__ shift,
    unsigned short* __restrict__ H, int M, int Nout, int Kp)
{
  size_t i = (size_t)blockIdx.x * 256 + threadIdx.x;
  if (i >= (size_t)M * Kp) return;
  int col = (int)(i % Kp);
  int row = (int)(i / Kp);
  float v = 0.f;
  if (col < Nout)
    v = fmaxf(Z[(size_t)row * Nout + col] * scale[col] + shift[col], 0.f);
  H[i] = f2bf(v);
}

// ---------------- 6) Final: out[n] = fm[n] + sum_j relu(bn(z3)) + bias ----------------
__global__ __launch_bounds__(256) void final_kernel(
    const float* __restrict__ Z, const float* __restrict__ scale, const float* __restrict__ shift,
    const float* __restrict__ fm, const float* __restrict__ bias,
    float* __restrict__ out, int M, int Nout)
{
  int wid  = (int)((blockIdx.x * 256 + threadIdx.x) >> 5);
  int lane = threadIdx.x & 31;
  if (wid >= M) return;
  float s = 0.f;
  for (int j = lane; j < Nout; j += 32)
    s += fmaxf(Z[(size_t)wid * Nout + j] * scale[j] + shift[j], 0.f);
  for (int off = 1; off < 32; off <<= 1) s += __shfl_xor(s, off, 32);
  if (lane == 0) out[wid] = fm[wid] + s + bias[0];
}

// ---------------- Launcher ----------------
extern "C" void kernel_launch(void* const* d_in, const int* in_sizes, int n_in,
                              void* d_out, int out_size, void* d_ws, size_t ws_size,
                              hipStream_t stream) {
  const int*   Xi   = (const int*)  d_in[0];
  const float* Xv   = (const float*)d_in[1];
  const float* emb1 = (const float*)d_in[2];
  const float* emb2 = (const float*)d_in[3];
  const float* W1 = (const float*)d_in[4],  *b1 = (const float*)d_in[5];
  const float* g1 = (const float*)d_in[6],  *bt1 = (const float*)d_in[7];
  const float* W2 = (const float*)d_in[8],  *b2 = (const float*)d_in[9];
  const float* g2 = (const float*)d_in[10], *bt2 = (const float*)d_in[11];
  const float* W3 = (const float*)d_in[12], *b3 = (const float*)d_in[13];
  const float* g3 = (const float*)d_in[14], *bt3 = (const float*)d_in[15];
  const float* bias = (const float*)d_in[16];
  float* out = (float*)d_out;

  char* ws = (char*)d_ws;
  // workspace layout (bytes)
  unsigned short* h0  = (unsigned short*)(ws);                         // 16384*640*2  = 20,971,520
  float*          Zb  = (float*)         (ws + 20971520);              // 16384*400*4  = 26,214,400
  unsigned short* h1  = (unsigned short*)(ws + 47185920);              // 16384*416*2  = 13,631,488
  float*          fm  = (float*)         (ws + 60817408);              // 16384*4      = 65,536
  float*          sc  = (float*)         (ws + 60882944);              // 400*4 (pad 2048)
  float*          sh  = (float*)         (ws + 60884992);              // 400*4 (pad 2048)
  unsigned short* W1b = (unsigned short*)(ws + 60887040);              // 400*640*2 (pad 524,288)
  unsigned short* W2b = (unsigned short*)(ws + 61411328);              // 400*416*2 (pad 335,872)
  unsigned short* W3b = (unsigned short*)(ws + 61747200);              // 400*416*2 (pad 335,872)
  unsigned short* h2  = h0;  // reuse h0 buffer for layer-3 input

  // weights -> bf16 padded
  convert_w_kernel<<<(HIDDEN * K1P + 255) / 256, 256, 0, stream>>>(W1, W1b, HIDDEN, K1, K1P);
  convert_w_kernel<<<(HIDDEN * K2P + 255) / 256, 256, 0, stream>>>(W2, W2b, HIDDEN, HIDDEN, K2P);
  convert_w_kernel<<<(HIDDEN * K2P + 255) / 256, 256, 0, stream>>>(W3, W3b, HIDDEN, HIDDEN, K2P);

  // embeddings + FM terms (one wave per row, 8 rows per block)
  embed_fm_kernel<<<NROWS / 8, 256, 0, stream>>>(Xi, Xv, emb1, emb2, h0, fm);

  // layer 1
  gemm_wmma_kernel<<<(NROWS / 16) * (HIDDEN / 16) / 8, 256, 0, stream>>>(h0, W1b, b1, Zb, NROWS, HIDDEN, K1P);
  bn_stats_kernel<<<HIDDEN, 256, 0, stream>>>(Zb, g1, bt1, sc, sh, NROWS, HIDDEN);
  bn_relu_bf16_kernel<<<(NROWS * K2P + 255) / 256, 256, 0, stream>>>(Zb, sc, sh, h1, NROWS, HIDDEN, K2P);

  // layer 2
  gemm_wmma_kernel<<<(NROWS / 16) * (HIDDEN / 16) / 8, 256, 0, stream>>>(h1, W2b, b2, Zb, NROWS, HIDDEN, K2P);
  bn_stats_kernel<<<HIDDEN, 256, 0, stream>>>(Zb, g2, bt2, sc, sh, NROWS, HIDDEN);
  bn_relu_bf16_kernel<<<(NROWS * K2P + 255) / 256, 256, 0, stream>>>(Zb, sc, sh, h2, NROWS, HIDDEN, K2P);

  // layer 3
  gemm_wmma_kernel<<<(NROWS / 16) * (HIDDEN / 16) / 8, 256, 0, stream>>>(h2, W3b, b3, Zb, NROWS, HIDDEN, K2P);
  bn_stats_kernel<<<HIDDEN, 256, 0, stream>>>(Zb, g3, bt3, sc, sh, NROWS, HIDDEN);

  // final reduction
  final_kernel<<<(NROWS * 32) / 256, 256, 0, stream>>>(Zb, sc, sh, fm, bias, out, NROWS, HIDDEN);
}
